// memory_11441792876847
// MI455X (gfx1250) — matrix operand order repaired and checked by reading, not compiled, not run on hardware
//
#include <hip/hip_runtime.h>
#include <hip/hip_bf16.h>
#include <math.h>

// ---------------------------------------------------------------------------
// Top-k via histogram threshold selection + two fused WMMA GEMM passes.
// MI455X / gfx1250: wave32, V_WMMA_F32_16X16X32_F16, async LDS staging.
//   s = q.mk^T + log(hist+beta)    (p_c normalizer cancels in the ratio)
//   pass1: GEMM -> per-row 512-bin histogram of s (range provably [-40,12])
//   thr  : scan bins top-down -> boundary bin edge + fractional occupancy
//   pass2: GEMM -> den/num with per-element weight sel in {1, frac, 0}
//   final: out = clip(num/den)
// Each wave processes TWO 16-row bands so every B fragment feeds 2 WMMAs,
// halving L2 traffic on the L2-resident mk_f16 (64MB in 192MB L2).
// ---------------------------------------------------------------------------

typedef __attribute__((ext_vector_type(16))) _Float16 v16h;
typedef __attribute__((ext_vector_type(8)))  _Float16 v8h;
typedef __attribute__((ext_vector_type(4)))  _Float16 v4h;
typedef __attribute__((ext_vector_type(8)))  float    v8f;

#define BB      1024
#define MM      131072
#define KK      256
#define RG      32                 // q-rows per block (2 bands per wave)
#define CH      8                  // M-chunks (grid.x)
#define WV      8                  // waves per block
#define NSL     (CH * WV)          // 64 slices along M
#define SLICE_M (MM / NSL)         // 2048 memory entries per wave
#define TILES   (SLICE_M / 16)     // 128 WMMA tiles per wave
#define TOPK    256
#define NBINS   512
#define NBW     (NBINS / 2)        // packed u16x2 words per row
#define SLO     (-40.0f)
#define SHI     (12.0f)
#define SSCALE  ((float)NBINS / (SHI - SLO))
#define BINW    ((SHI - SLO) / (float)NBINS)
#define EPSF    1e-3f
#define BETAF   1e-8f

#if __has_builtin(__builtin_amdgcn_global_load_async_to_lds_b128) && \
    __has_builtin(__builtin_amdgcn_s_wait_asynccnt)
#define HAVE_ASYNC 1
// Builtin signature (from hipcc diagnostic): arg0 = int4 addrspace(1)*,
// then LDS ptr, imm offset, imm cpol.
typedef __attribute__((__vector_size__(4 * sizeof(int)))) int v4i;
typedef __attribute__((address_space(1))) v4i* gptr_v4i;
typedef __attribute__((address_space(3))) v4i* lptr_v4i;
#else
#define HAVE_ASYNC 0
#endif

// ---------------- fragment helpers -----------------------------------------
__device__ __forceinline__ v16h load_bfrag(const _Float16* p) {
  v8h lo = *(const v8h*)(p);
  v8h up = *(const v8h*)(p + 8);
  v16h b;
#pragma unroll
  for (int i = 0; i < 8; ++i) { b[i] = lo[i]; b[8 + i] = up[i]; }
  return b;
}

// A fragment, contiguous k order (B uses the same): lane L(0-15)=row L; K runs [kb,kb+16),
// kb = kk*32 + (lane>=16 ? 16 : 0).
__device__ __forceinline__ void load_afrag_g(v16h a[8], const _Float16* qrow) {
#pragma unroll
  for (int kk = 0; kk < 8; ++kk) {
    v8h lo = *(const v8h*)(qrow + kk * 32);
    v8h up = *(const v8h*)(qrow + kk * 32 + 8);
#pragma unroll
    for (int i = 0; i < 8; ++i) { a[kk][i] = lo[i]; a[kk][8 + i] = up[i]; }
  }
}

// One 16x16 B tile feeding both bands: 8 k-steps, ping-pong B, 2 acc chains.
__device__ __forceinline__ void tile_gemm2(const v16h a0[8], const v16h a1[8],
                                           const _Float16* brow,
                                           v8f& o0, v8f& o1) {
  v8f acc0 = {}, acc1 = {};
  v16h bf0 = load_bfrag(brow), bf1;
#pragma unroll
  for (int kk = 0; kk < 8; ++kk) {
    v16h cur = (kk & 1) ? bf1 : bf0;
    if (kk < 7) {
      if (kk & 1) bf0 = load_bfrag(brow + (kk + 1) * 32);
      else        bf1 = load_bfrag(brow + (kk + 1) * 32);
    }
    acc0 = __builtin_amdgcn_wmma_f32_16x16x32_f16(false, a0[kk], false, cur,
                                                  (short)0, acc0, false, false);
    acc1 = __builtin_amdgcn_wmma_f32_16x16x32_f16(false, a1[kk], false, cur,
                                                  (short)0, acc1, false, false);
  }
  o0 = acc0; o1 = acc1;
}

// ---- prep: f32 -> f16 for q and memory_key, log(hist + beta) ---------------
__global__ __launch_bounds__(256) void prep_kernel(
    const float* __restrict__ q, const float* __restrict__ mk,
    const float* __restrict__ hist,
    _Float16* __restrict__ qh, _Float16* __restrict__ mkh,
    float* __restrict__ lpc) {
  const int id = blockIdx.x * blockDim.x + threadIdx.x;   // covers M*K/4
  const int i4 = id * 4;
  if (i4 < MM * KK) {
    float4 v = ((const float4*)mk)[id];
    v4h h; h.x = (_Float16)v.x; h.y = (_Float16)v.y;
           h.z = (_Float16)v.z; h.w = (_Float16)v.w;
    ((v4h*)mkh)[id] = h;
  }
  if (i4 < BB * KK) {
    float4 v = ((const float4*)q)[id];
    v4h h; h.x = (_Float16)v.x; h.y = (_Float16)v.y;
           h.z = (_Float16)v.z; h.w = (_Float16)v.w;
    ((v4h*)qh)[id] = h;
  }
  if (i4 < MM) {
    float4 v = ((const float4*)hist)[id];
    float4 o; o.x = __logf(v.x + BETAF); o.y = __logf(v.y + BETAF);
              o.z = __logf(v.z + BETAF); o.w = __logf(v.w + BETAF);
    ((float4*)lpc)[id] = o;
  }
}

// ---- zero the global histogram (ws is not re-poisoned between replays) ----
__global__ __launch_bounds__(256) void zero_kernel(uint4* __restrict__ p, int n4) {
  const int id = blockIdx.x * blockDim.x + threadIdx.x;
  if (id < n4) { uint4 z = {0u, 0u, 0u, 0u}; p[id] = z; }
}

// ---- pass 1: GEMM + per-row score histogram (u16x2-packed LDS bins) --------
__global__ __launch_bounds__(256) void hist_kernel(
    const _Float16* __restrict__ qh, const _Float16* __restrict__ mkh,
    const float* __restrict__ lpc, unsigned int* __restrict__ histg) {
  __shared__ unsigned int lh[RG * NBW];            // 32 KB
#if HAVE_ASYNC
  __shared__ __align__(16) _Float16 lds_a[RG * KK];  // 16 KB q-band stage
#endif
  const int tid = threadIdx.x;
  const int rg  = blockIdx.y;
  for (int i = tid; i < RG * NBW; i += 256) lh[i] = 0u;
#if HAVE_ASYNC
  {
    // Stage the block-shared 32x256 f16 q-band into LDS asynchronously.
    const _Float16* gsrc = qh + (size_t)rg * RG * KK;
#pragma unroll
    for (int j = 0; j < (RG * KK * 2 / 16) / 256; ++j) {   // 4 x b128 / thread
      const int byte_off = (tid + j * 256) * 16;
      __builtin_amdgcn_global_load_async_to_lds_b128(
          (gptr_v4i)((char*)gsrc + byte_off),
          (lptr_v4i)((char*)lds_a + byte_off), 0, 0);
    }
    __builtin_amdgcn_s_wait_asynccnt(0);
  }
#endif
  __syncthreads();

  const int lane   = tid & 31;
  const int wave   = tid >> 5;
  const int slice  = blockIdx.x * WV + wave;
  const int base_m = slice * SLICE_M;
  const int lane16 = lane & 15;
  const int hi     = lane >> 4;
  const int kb_hi  = hi << 4;

  v16h a0[8], a1[8];
#if HAVE_ASYNC
  load_afrag_g(a0, (const _Float16*)lds_a + (size_t)lane16 * KK + hi * 16);
  load_afrag_g(a1, (const _Float16*)lds_a + (size_t)(16 + lane16) * KK + hi * 16);
#else
  load_afrag_g(a0, qh + (size_t)(rg * RG + lane16) * KK + hi * 16);
  load_afrag_g(a1, qh + (size_t)(rg * RG + 16 + lane16) * KK + hi * 16);
#endif

  for (int t = 0; t < TILES; ++t) {
    const int col = base_m + t * 16 + lane16;
    const _Float16* brow = mkh + (size_t)col * KK + kb_hi;
    if (t + 1 < TILES) __builtin_prefetch(brow + (size_t)16 * KK, 0, 1);
    v8f acc0, acc1;
    tile_gemm2(a0, a1, brow, acc0, acc1);
    const float lp = lpc[col];
#pragma unroll
    for (int b = 0; b < 2; ++b) {
      v8f acc = b ? acc1 : acc0;
#pragma unroll
      for (int r = 0; r < 8; ++r) {
        float s = acc[r] + lp;
        int bi = (int)((s - SLO) * SSCALE);
        bi = bi < 0 ? 0 : (bi > NBINS - 1 ? NBINS - 1 : bi);
        const int rowl = b * 16 + hi * 8 + r;
        atomicAdd(&lh[rowl * NBW + (bi >> 1)], 1u << ((bi & 1) * 16));
      }
    }
  }
  __syncthreads();

  // Merge into global per-row histogram (integer atomics: deterministic).
  for (int i = tid; i < RG * NBW; i += 256) {
    unsigned int w = lh[i];
    if (!w) continue;
    const size_t row = (size_t)rg * RG + (i / NBW);
    const int word = i % NBW;
    unsigned int c0 = w & 0xffffu, c1 = w >> 16;
    if (c0) atomicAdd(&histg[row * NBINS + 2 * word], c0);
    if (c1) atomicAdd(&histg[row * NBINS + 2 * word + 1], c1);
  }
}

// ---- threshold: scan bins top-down -> (upper edge, fractional occupancy) ---
__global__ __launch_bounds__(256) void thr_kernel(
    const unsigned int* __restrict__ histg, float2* __restrict__ thr) {
  const int row = blockIdx.x * blockDim.x + threadIdx.x;
  if (row >= BB) return;
  const unsigned int* h = histg + (size_t)row * NBINS;
  unsigned int cum = 0;
  for (int b = NBINS - 1; b >= 0; --b) {
    unsigned int c = h[b];
    if (cum + c >= TOPK) {
      float2 tv;
      tv.x = SLO + (float)(b + 1) * BINW;                // fully-in above this
      tv.y = (float)(TOPK - cum) / (float)c;             // boundary-bin frac
      thr[row] = tv;
      return;
    }
    cum += c;
  }
  float2 tv; tv.x = -1e30f; tv.y = 0.f;                  // include everything
  thr[row] = tv;
}

// ---- pass 2: GEMM + thresholded weight accumulation ------------------------
__global__ __launch_bounds__(256) void accum_kernel(
    const _Float16* __restrict__ qh, const _Float16* __restrict__ mkh,
    const float* __restrict__ vals, const float* __restrict__ lpc,
    const float2* __restrict__ thr, float2* __restrict__ partials) {
#if HAVE_ASYNC
  __shared__ __align__(16) _Float16 lds_a[RG * KK];
#endif
  const int tid = threadIdx.x;
  const int rg  = blockIdx.y;
#if HAVE_ASYNC
  {
    const _Float16* gsrc = qh + (size_t)rg * RG * KK;
#pragma unroll
    for (int j = 0; j < (RG * KK * 2 / 16) / 256; ++j) {
      const int byte_off = (tid + j * 256) * 16;
      __builtin_amdgcn_global_load_async_to_lds_b128(
          (gptr_v4i)((char*)gsrc + byte_off),
          (lptr_v4i)((char*)lds_a + byte_off), 0, 0);
    }
    __builtin_amdgcn_s_wait_asynccnt(0);
  }
  __syncthreads();
#endif

  const int lane   = tid & 31;
  const int wave   = tid >> 5;
  const int slice  = blockIdx.x * WV + wave;
  const int base_m = slice * SLICE_M;
  const int lane16 = lane & 15;
  const int hi     = lane >> 4;
  const int kb_hi  = hi << 4;

  v16h a0[8], a1[8];
#if HAVE_ASYNC
  load_afrag_g(a0, (const _Float16*)lds_a + (size_t)lane16 * KK + hi * 16);
  load_afrag_g(a1, (const _Float16*)lds_a + (size_t)(16 + lane16) * KK + hi * 16);
#else
  load_afrag_g(a0, qh + (size_t)(rg * RG + lane16) * KK + hi * 16);
  load_afrag_g(a1, qh + (size_t)(rg * RG + 16 + lane16) * KK + hi * 16);
#endif

  float th[2][8], fr[2][8], den[2][8], num[2][8];
#pragma unroll
  for (int b = 0; b < 2; ++b)
#pragma unroll
    for (int r = 0; r < 8; ++r) {
      float2 tv = thr[rg * RG + b * 16 + hi * 8 + r];
      th[b][r] = tv.x; fr[b][r] = tv.y;
      den[b][r] = 0.f; num[b][r] = 0.f;
    }

  for (int t = 0; t < TILES; ++t) {
    const int col = base_m + t * 16 + lane16;
    const _Float16* brow = mkh + (size_t)col * KK + kb_hi;
    if (t + 1 < TILES) __builtin_prefetch(brow + (size_t)16 * KK, 0, 1);
    v8f acc0, acc1;
    tile_gemm2(a0, a1, brow, acc0, acc1);
    const float lp = lpc[col];
    const float vv = vals[col];
#pragma unroll
    for (int b = 0; b < 2; ++b) {
      v8f acc = b ? acc1 : acc0;
#pragma unroll
      for (int r = 0; r < 8; ++r) {
        float s = acc[r] + lp;
        float sel = s >= th[b][r] ? 1.f
                  : (s >= th[b][r] - BINW ? fr[b][r] : 0.f);
        float w = __expf(s) * sel;      // s in [-40,12]: exp never overflows
        den[b][r] += w;
        num[b][r] += vv * w;
      }
    }
  }

  // Tree-add across the 16 lanes sharing the same rows (fixed order).
#pragma unroll
  for (int off = 1; off < 16; off <<= 1) {
#pragma unroll
    for (int b = 0; b < 2; ++b)
#pragma unroll
      for (int r = 0; r < 8; ++r) {
        den[b][r] += __shfl_xor(den[b][r], off, 32);
        num[b][r] += __shfl_xor(num[b][r], off, 32);
      }
  }

  if (lane16 == 0) {   // lane 0 -> rows +0..7, lane 16 -> rows +8..15
#pragma unroll
    for (int b = 0; b < 2; ++b)
#pragma unroll
      for (int r = 0; r < 8; ++r) {
        const int rowg = rg * RG + b * 16 + hi * 8 + r;
        float2 p; p.x = den[b][r]; p.y = num[b][r];
        partials[(size_t)rowg * NSL + slice] = p;
      }
  }
}

// ---- finalize: reduce slices in fixed order, ratio + clip ------------------
__global__ __launch_bounds__(256) void finalize_kernel(
    const float2* __restrict__ partials, float* __restrict__ out) {
  const int lane = threadIdx.x & 31;
  const int row  = blockIdx.x * 8 + (threadIdx.x >> 5);
  if (row >= BB) return;
  float den = 0.f, num = 0.f;
#pragma unroll
  for (int j = 0; j < NSL / 32; ++j) {
    float2 p = partials[(size_t)row * NSL + (j * 32 + lane)];
    den += p.x; num += p.y;
  }
#pragma unroll
  for (int off = 1; off < 32; off <<= 1) {
    den += __shfl_xor(den, off, 32);
    num += __shfl_xor(num, off, 32);
  }
  if (lane == 0)
    out[row] = fminf(fmaxf(num / den, EPSF), 1.0f - EPSF);
}

// ---------------------------------------------------------------------------
extern "C" void kernel_launch(void* const* d_in, const int* in_sizes, int n_in,
                              void* d_out, int out_size, void* d_ws, size_t ws_size,
                              hipStream_t stream) {
  const float* q    = (const float*)d_in[0];   // [B,K]
  const float* mk   = (const float*)d_in[1];   // [M,K]
  const float* vals = (const float*)d_in[2];   // [M]
  const float* hist = (const float*)d_in[3];   // [M]
  float* out = (float*)d_out;

  // Workspace layout (16B aligned). mk_f16 (64MB) is L2-resident (192MB L2).
  char* ws = (char*)d_ws;
  _Float16*     mkh   = (_Float16*)(ws);                         // 64 MB
  _Float16*     qh    = (_Float16*)(ws + (size_t)67108864);      // 512 KB
  float*        lpc   = (float*)   (ws + (size_t)67633152);      // 512 KB
  unsigned int* histg = (unsigned int*)(ws + (size_t)68157440);  // 2 MB
  float2*       thr   = (float2*)  (ws + (size_t)70254592);      // 8 KB
  float2*       parts = (float2*)  (ws + (size_t)70262784);      // 512 KB

  prep_kernel<<<(MM * KK / 4) / 256, 256, 0, stream>>>(q, mk, hist, qh, mkh, lpc);
  zero_kernel<<<(BB * NBINS / 4 + 255) / 256, 256, 0, stream>>>(
      (uint4*)histg, BB * NBINS / 4);

  dim3 grid(CH, BB / RG);   // 8 M-chunks x 32 row-groups, 8 waves each
  hist_kernel<<<grid, 256, 0, stream>>>(qh, mkh, lpc, histg);
  thr_kernel<<<BB / 256, 256, 0, stream>>>(histg, thr);
  accum_kernel<<<grid, 256, 0, stream>>>(qh, mkh, vals, lpc, thr, parts);
  finalize_kernel<<<BB / 8, 256, 0, stream>>>(parts, out);
}
